// ControlPointNet_fail_25555055411860
// MI455X (gfx1250) — matrix operand order repaired and checked by reading, not compiled, not run on hardware
//
#include <hip/hip_runtime.h>
#include <hip/hip_bf16.h>

typedef __attribute__((ext_vector_type(16))) _Float16 v16h;
typedef __attribute__((ext_vector_type(8)))  _Float16 v8h;
typedef __attribute__((ext_vector_type(2)))  _Float16 v2h;
typedef __attribute__((ext_vector_type(8)))  float    v8f;

#define NB 2
#define DD 96
#define HH 96
#define WW 96
#define HWC (HH*WW)          // 9216
#define VOX (DD*HH*WW)       // 884736
#define KTOP 1000
#define CAND_CAP 4096
#define SPT (NB*VOX/16)      // 110592 spatial tiles of 16

// ---------------------------------------------------------------------------
// Weight packing: fp32 [Co][Ci][3][3][3] -> f16 WMMA A-fragments with the
// K axis reordered as k = t*Ci + ci  (t = kd*9+kh*3+kw), matching the
// channel-last B gather. A-fragment k order used by this code (not ISA table order):
// lane row M = lane&15; element e -> K_in_chunk = (lane>>4)*16 + e.
// ---------------------------------------------------------------------------
__global__ void pack_weights(const float* __restrict__ w, _Float16* __restrict__ wp,
                             int Ci, int Co, int ks3, int nkc)
{
    const int lane = threadIdx.x;      // 32
    const int kc = blockIdx.x;         // k-chunk
    const int cot = blockIdx.y;        // co tile
    const int m = lane & 15, hi = lane >> 4;
    const int co = cot * 16 + m;
    const int Ktot = Ci * ks3;
    _Float16* dst = wp + ((size_t)(cot * nkc + kc) * 32 + lane) * 16;
#pragma unroll
    for (int e = 0; e < 16; ++e) {
        int k = kc * 32 + hi * 16 + e;
        float val = 0.f;
        if (co < Co && k < Ktot) {
            int t = k / Ci, ci = k - t * Ci;
            val = w[((size_t)co * Ci + ci) * ks3 + t];
        }
        dst[e] = (_Float16)val;
    }
}

// ---------------------------------------------------------------------------
// Implicit-GEMM conv3d via V_WMMA_F32_16X16X32_F16, channel-last activations.
// Fully static shape: one wave computes ALL co-tiles (COT WMMAs per B build)
// for 16 spatial positions; K-loop in NKC chunks of 32.
// B k order used here (same as A): lane N = lane&15, element e -> K = (lane>>4)*16 + e.
// With k = t*CI + ci, each SEG=min(CI,16) consecutive elements are contiguous
// ci at one spatial offset -> single wide load per segment.
// ---------------------------------------------------------------------------
template<int KS, int CI, int CO>
__global__ void conv_wmma(const _Float16* __restrict__ in,    // [NB*VOX*CI] ch-last
                          const _Float16* __restrict__ wpack, // packed A frags
                          _Float16* __restrict__ out)         // [NB*VOX*CO] ch-last
{
    constexpr int PAD  = (KS == 3) ? 1 : 0;
    constexpr int Ktot = KS * KS * KS * CI;
    constexpr int NKC  = (Ktot + 31) / 32;
    constexpr int SEG  = (CI < 16) ? CI : 16;
    constexpr int NSEG = 16 / SEG;
    constexpr int COT  = (CO + 15) / 16;

    const int lane = threadIdx.x;
    const int n  = lane & 15;
    const int hi = lane >> 4;
    const int p  = blockIdx.x * 16 + n;    // global spatial index over NB*VOX
    const int nb = p / VOX;
    const int v  = p - nb * VOX;
    const int d  = v / HWC;
    const int hw = v - d * HWC;
    const int h  = hw / WW;
    const int w  = hw - h * WW;
    const _Float16* inb = in + (size_t)nb * VOX * CI;

    v8f c0 = {}, c1 = {};
    for (int kc = 0; kc < NKC; ++kc) {
        v16h a0 = *(const v16h*)(wpack + ((size_t)kc * 32 + lane) * 16);
        if (kc + 1 < NKC)   // gfx1250 global_prefetch_b8: next chunk's A stream
            __builtin_prefetch(wpack + ((size_t)((COT - 1) * NKC + kc + 1) * 32 + lane) * 16,
                               0, 1);
        v16h b;
#pragma unroll
        for (int e = 0; e < 16; ++e) b[e] = (_Float16)0.f;
        const int kbase = kc * 32 + hi * 16;
#pragma unroll
        for (int sgi = 0; sgi < NSEG; ++sgi) {
            const int k0  = kbase + sgi * SEG;   // multiple of SEG
            const int t   = k0 / CI;             // CI is a power of two
            const int ci0 = k0 - t * CI;
            bool ok = (k0 < Ktot);
            int kd, kh, kw;
            if (KS == 3) { kd = t / 9; int r = t - kd * 9; kh = r / 3; kw = r - kh * 3; }
            else         { kd = 0; kh = 0; kw = 0; }
            const int zd = d + kd - PAD, zh = h + kh - PAD, zw = w + kw - PAD;
            ok = ok && (unsigned)zd < (unsigned)DD && (unsigned)zh < (unsigned)HH &&
                       (unsigned)zw < (unsigned)WW;
            if (ok) {
                const _Float16* src = inb + (size_t)(zd * HWC + zh * WW + zw) * CI + ci0;
                if constexpr (SEG == 16) {
                    v16h tv = *(const v16h*)src;
#pragma unroll
                    for (int e = 0; e < 16; ++e) b[sgi * 16 + e] = tv[e];
                } else if constexpr (SEG == 8) {
                    v8h tv = *(const v8h*)src;
#pragma unroll
                    for (int e = 0; e < 8; ++e) b[sgi * 8 + e] = tv[e];
                } else {
                    v2h tv = *(const v2h*)src;
#pragma unroll
                    for (int e = 0; e < 2; ++e) b[sgi * 2 + e] = tv[e];
                }
            }
        }
        c0 = __builtin_amdgcn_wmma_f32_16x16x32_f16(false, a0, false, b, (short)0, c0,
                                                    false, false);
        if constexpr (COT == 2) {
            v16h a1 = *(const v16h*)(wpack + ((size_t)(NKC + kc) * 32 + lane) * 16);
            c1 = __builtin_amdgcn_wmma_f32_16x16x32_f16(false, a1, false, b, (short)0, c1,
                                                        false, false);
        }
    }
    // C/D 16x16 f32 layout: lane N = lane&15, VGPR r -> M = r + 8*(lane>>4).
    // Channel-last store: co = cotile*16 + hi*8 + r is contiguous in r.
    const int co0 = hi * 8;
    _Float16* op = out + (size_t)p * CO + co0;
#pragma unroll
    for (int r = 0; r < 8; ++r)
        if (co0 + r < CO) op[r] = (_Float16)c0[r];
    if constexpr (COT == 2) {
        _Float16* op1 = op + 16;
#pragma unroll
        for (int r = 0; r < 8; ++r) op1[r] = (_Float16)c1[r];
    }
}

// ---------------------------------------------------------------------------
// Elementwise / reduction helpers (channel-last activations)
// ---------------------------------------------------------------------------
// x fp32 [2,2,V] channel-major -> f16 channel-last [2,V,2]
__global__ void input_to_f16(const float* __restrict__ in, _Float16* __restrict__ out)
{
    size_t i = (size_t)blockIdx.x * 256 + threadIdx.x;   // NB*VOX*2
    int nb = (int)(i / ((size_t)VOX * 2));
    int r  = (int)(i - (size_t)nb * VOX * 2);
    int v = r >> 1, ci = r & 1;
    out[i] = (_Float16)in[((size_t)nb * 2 + ci) * VOX + v];
}

__global__ void zero_f32(float* p, int n)
{
    int i = blockIdx.x * 256 + threadIdx.x;
    if (i < n) p[i] = 0.f;
}

// channel-last stats: C divides 256, so each thread's channel is fixed.
__global__ void bn_stats(const _Float16* __restrict__ x, float* sum, float* ssq, int C)
{
    __shared__ float ls[256], lq[256];
    const int tid = threadIdx.x;
    size_t base = (size_t)blockIdx.x * 8192;
    float s = 0.f, q = 0.f;
#pragma unroll 4
    for (int j = 0; j < 32; ++j) {
        float t = (float)x[base + tid + j * 256];
        s += t; q += t * t;
    }
    ls[tid] = s; lq[tid] = q;
    __syncthreads();
    for (int off = 128; off >= C; off >>= 1) {   // off % C == 0 throughout
        if (tid < off) { ls[tid] += ls[tid + off]; lq[tid] += lq[tid + off]; }
        __syncthreads();
    }
    if (tid < C) { atomicAdd(&sum[tid], ls[tid]); atomicAdd(&ssq[tid], lq[tid]); }
}

__global__ void bn_final(const float* sum, const float* ssq, const float* g,
                         const float* b, float* scale, float* shift, int C)
{
    int c = threadIdx.x;
    if (c >= C) return;
    float cnt = (float)NB * (float)VOX;
    float m = sum[c] / cnt;
    float var = ssq[c] / cnt - m * m;
    float sc = g[c] * rsqrtf(var + 1e-5f);
    scale[c] = sc;
    shift[c] = b[c] - m * sc;
}

__global__ void bn_relu(const _Float16* __restrict__ x, _Float16* __restrict__ y,
                        const float* scale, const float* shift, int C)
{
    size_t i = (size_t)blockIdx.x * 256 + threadIdx.x;   // NB*VOX*C total
    int c = (int)(i % C);
    float t = (float)x[i] * scale[c] + shift[c];
    y[i] = (_Float16)fmaxf(t, 0.f);
}

__global__ void resid_relu(const _Float16* __restrict__ a, const _Float16* __restrict__ b,
                           _Float16* __restrict__ y,
                           const float* scA, const float* shA,
                           const float* scB, const float* shB, int C)
{
    size_t i = (size_t)blockIdx.x * 256 + threadIdx.x;
    int c = (int)(i % C);
    float t = (float)a[i] * scA[c] + shA[c] + (float)b[i] * scB[c] + shB[c];
    y[i] = (_Float16)fmaxf(t, 0.f);
}

// 1x1x1 head on channel-last acts (contiguous 32-dot); sigmoid omitted
// (monotonic -> identical top-k ranking).
__global__ void score_kernel(const _Float16* __restrict__ act,
                             const float* __restrict__ w3, const float* __restrict__ b3,
                             float* __restrict__ sc)
{
    size_t i = (size_t)blockIdx.x * 256 + threadIdx.x;   // NB*VOX
    const _Float16* p = act + i * 32;
    float s = b3[0];
#pragma unroll
    for (int c = 0; c < 32; ++c) s += (float)p[c] * w3[c];
    sc[i] = s;
}

// ---------------------------------------------------------------------------
// Top-k(1000) per batch: radix-select + collect + bitonic sort + emit coords
// ---------------------------------------------------------------------------
__device__ __forceinline__ unsigned mono_key(float f)
{
    unsigned u = __float_as_uint(f);
    return (u & 0x80000000u) ? ~u : (u | 0x80000000u);
}

__global__ void topk_init(unsigned* prefix, unsigned* remk, unsigned* candcnt,
                          unsigned* hist, unsigned long long* cand)
{
    int i = blockIdx.x * 256 + threadIdx.x;
    if (i < NB) { prefix[i] = 0; remk[i] = KTOP; candcnt[i] = 0; }
    if (i < NB * 256) hist[i] = 0;
    for (int j = i; j < NB * CAND_CAP; j += gridDim.x * 256) cand[j] = 0ull;
}

__global__ void topk_hist(const float* __restrict__ sc, const unsigned* prefix,
                          unsigned* hist, int pass)
{
    __shared__ unsigned lh[256];
    const int tid = threadIdx.x;
    const int nb = blockIdx.y;
    lh[tid] = 0;
    __syncthreads();
    const int shift = 24 - 8 * pass;
    const unsigned pmask = pass ? (0xFFFFFFFFu << (shift + 8)) : 0u;
    const unsigned pref = prefix[nb];
    size_t base = (size_t)nb * VOX + (size_t)blockIdx.x * 4096;
#pragma unroll 4
    for (int j = 0; j < 16; ++j) {
        unsigned mk = mono_key(sc[base + tid + j * 256]);
        if ((mk & pmask) == pref) atomicAdd(&lh[(mk >> shift) & 0xFF], 1u);
    }
    __syncthreads();
    if (lh[tid]) atomicAdd(&hist[nb * 256 + tid], lh[tid]);
}

__global__ void topk_select(unsigned* prefix, unsigned* remk, unsigned* hist, int pass)
{
    __shared__ unsigned h[256];
    const int nb = blockIdx.x, tid = threadIdx.x;
    h[tid] = hist[nb * 256 + tid];
    __syncthreads();
    if (tid == 0) {
        const int shift = 24 - 8 * pass;
        unsigned rk = remk[nb], acc = 0, chosen = 0;
        for (int b = 255; b >= 0; --b) {
            if (acc + h[b] >= rk) { chosen = (unsigned)b; break; }
            acc += h[b];
        }
        remk[nb] = rk - acc;
        prefix[nb] |= chosen << shift;
    }
    __syncthreads();
    hist[nb * 256 + tid] = 0;   // ready for next pass
}

__global__ void topk_collect(const float* __restrict__ sc, const unsigned* prefix,
                             unsigned* candcnt, unsigned long long* cand)
{
    const int nb = blockIdx.y;
    const unsigned T = prefix[nb];
#pragma unroll 4
    for (int j = 0; j < 16; ++j) {
        unsigned e = blockIdx.x * 4096 + threadIdx.x + j * 256;
        unsigned mk = mono_key(sc[(size_t)nb * VOX + e]);
        if (mk >= T) {
            unsigned pos = atomicAdd(&candcnt[nb], 1u);
            if (pos < CAND_CAP)
                cand[(size_t)nb * CAND_CAP + pos] =
                    ((unsigned long long)mk << 32) | (unsigned)(~e);
        }
    }
}

__global__ __launch_bounds__(1024) void topk_sort(unsigned long long* cand)
{
    __shared__ unsigned long long s[CAND_CAP];
    const int nb = blockIdx.x, tid = threadIdx.x;
    for (int i = tid; i < CAND_CAP; i += 1024) s[i] = cand[(size_t)nb * CAND_CAP + i];
    __syncthreads();
    for (int k = 2; k <= CAND_CAP; k <<= 1) {
        for (int j = k >> 1; j > 0; j >>= 1) {
            for (int i = tid; i < CAND_CAP; i += 1024) {
                int ixj = i ^ j;
                if (ixj > i) {
                    bool up = ((i & k) == 0);     // descending overall
                    unsigned long long a = s[i], b = s[ixj];
                    if (up ? (a < b) : (a > b)) { s[i] = b; s[ixj] = a; }
                }
            }
            __syncthreads();
        }
    }
    for (int i = tid; i < CAND_CAP; i += 1024) cand[(size_t)nb * CAND_CAP + i] = s[i];
}

__global__ void topk_emit(const unsigned long long* __restrict__ cand,
                          float* __restrict__ out)
{
    const int nb = blockIdx.x;
    const int j = blockIdx.y * 256 + threadIdx.x;
    if (j >= KTOP) return;
    unsigned e = ~(unsigned)(cand[(size_t)nb * CAND_CAP + j] & 0xFFFFFFFFull);
    int d = e / HWC; int r = e - d * HWC; int h = r / WW; int w = r - h * WW;
    const float st = 2.0f / 95.0f;
    out[nb * 3 * KTOP + j]            = -1.f + st * (float)d;
    out[nb * 3 * KTOP + KTOP + j]     = -1.f + st * (float)h;
    out[nb * 3 * KTOP + 2 * KTOP + j] = -1.f + st * (float)w;
}

// ---------------------------------------------------------------------------
// Orchestration
// ---------------------------------------------------------------------------
extern "C" void kernel_launch(void* const* d_in, const int* in_sizes, int n_in,
                              void* d_out, int out_size, void* d_ws, size_t ws_size,
                              hipStream_t stream)
{
    (void)in_sizes; (void)n_in; (void)out_size; (void)ws_size;
    char* ws = (char*)d_ws;
    const size_t BUF = (size_t)NB * 32 * VOX * sizeof(_Float16);  // 113 MB
    _Float16* t1 = (_Float16*)(ws);
    _Float16* t2 = (_Float16*)(ws + BUF);
    _Float16* P  = (_Float16*)(ws + 2 * BUF);
    _Float16* Q  = (_Float16*)(ws + 3 * BUF);
    size_t o = 4 * BUF;

    float* stats = (float*)(ws + o); o += 1024;
    float *sumA = stats,       *ssqA = stats + 32,  *scA = stats + 64,  *shA = stats + 96;
    float *sumB = stats + 128, *ssqB = stats + 160, *scB = stats + 192, *shB = stats + 224;

    unsigned* prefix  = (unsigned*)(ws + o); o += 256;
    unsigned* remk    = prefix + 8;
    unsigned* candcnt = prefix + 16;
    unsigned* hist = (unsigned*)(ws + o); o += NB * 256 * 4;
    o = (o + 255) & ~(size_t)255;
    unsigned long long* cand = (unsigned long long*)(ws + o); o += NB * CAND_CAP * 8;
    o = (o + 255) & ~(size_t)255;

    // conv configs: per layer {w1, w2, ws}
    struct CC { int wi, Ci, Co, ks; };
    const CC cfg[9] = {
        {1, 2, 8, 3},  {4, 8, 8, 3},   {7, 2, 8, 1},
        {10, 8, 16, 3},{13, 16, 16, 3},{16, 8, 16, 1},
        {19, 16, 32, 3},{22, 32, 32, 3},{25, 16, 32, 1}};
    _Float16* wp[9];
    for (int j = 0; j < 9; ++j) {
        int ks3 = cfg[j].ks * cfg[j].ks * cfg[j].ks;
        int nkc = (cfg[j].Ci * ks3 + 31) / 32;
        int cot = (cfg[j].Co + 15) / 16;
        wp[j] = (_Float16*)(ws + o);
        o += (size_t)cot * nkc * 1024;
        o = (o + 255) & ~(size_t)255;
        pack_weights<<<dim3(nkc, cot), 32, 0, stream>>>(
            (const float*)d_in[cfg[j].wi], wp[j], cfg[j].Ci, cfg[j].Co, ks3, nkc);
    }

    // input fp32 [2,2,V] -> f16 channel-last [2,V,2] in P
    input_to_f16<<<(unsigned)(((size_t)NB * VOX * 2 + 255) / 256), 256, 0, stream>>>(
        (const float*)d_in[0], P);

    for (int L = 0; L < 3; ++L) {
        const int base = 1 + 9 * L;
        const float* g1 = (const float*)d_in[base + 1];
        const float* b1 = (const float*)d_in[base + 2];
        const float* g2 = (const float*)d_in[base + 4];
        const float* b2 = (const float*)d_in[base + 5];
        const float* gs = (const float*)d_in[base + 7];
        const float* bs = (const float*)d_in[base + 8];
        const int j1 = L * 3, j2 = L * 3 + 1, js = L * 3 + 2;
        const int Co = (L == 0) ? 8 : (L == 1) ? 16 : 32;
        const size_t nel = (size_t)NB * Co * VOX;
        const unsigned nb256 = (unsigned)(nel / 256);
        const unsigned nstat = (unsigned)(nel / 8192);

        // t1 = conv1(P); Q = relu(bn1(t1))
        if (L == 0)      conv_wmma<3, 2,  8><<<SPT, 32, 0, stream>>>(P, wp[j1], t1);
        else if (L == 1) conv_wmma<3, 8,  16><<<SPT, 32, 0, stream>>>(P, wp[j1], t1);
        else             conv_wmma<3, 16, 32><<<SPT, 32, 0, stream>>>(P, wp[j1], t1);
        zero_f32<<<1, 256, 0, stream>>>(stats, 256);
        bn_stats<<<nstat, 256, 0, stream>>>(t1, sumA, ssqA, Co);
        bn_final<<<1, Co, 0, stream>>>(sumA, ssqA, g1, b1, scA, shA, Co);
        bn_relu<<<nb256, 256, 0, stream>>>(t1, Q, scA, shA, Co);
        // t2 = conv2(Q); t1 = conv_s(P)  (t1's old contents fully consumed)
        if (L == 0)      conv_wmma<3, 8,  8><<<SPT, 32, 0, stream>>>(Q, wp[j2], t2);
        else if (L == 1) conv_wmma<3, 16, 16><<<SPT, 32, 0, stream>>>(Q, wp[j2], t2);
        else             conv_wmma<3, 32, 32><<<SPT, 32, 0, stream>>>(Q, wp[j2], t2);
        if (L == 0)      conv_wmma<1, 2,  8><<<SPT, 32, 0, stream>>>(P, wp[js], t1);
        else if (L == 1) conv_wmma<1, 8,  16><<<SPT, 32, 0, stream>>>(P, wp[js], t1);
        else             conv_wmma<1, 16, 32><<<SPT, 32, 0, stream>>>(P, wp[js], t1);
        // P = relu(bn2(t2) + bns(t1))
        zero_f32<<<1, 256, 0, stream>>>(stats, 256);
        bn_stats<<<nstat, 256, 0, stream>>>(t2, sumA, ssqA, Co);
        bn_stats<<<nstat, 256, 0, stream>>>(t1, sumB, ssqB, Co);
        bn_final<<<1, Co, 0, stream>>>(sumA, ssqA, g2, b2, scA, shA, Co);
        bn_final<<<1, Co, 0, stream>>>(sumB, ssqB, gs, bs, scB, shB, Co);
        resid_relu<<<nb256, 256, 0, stream>>>(t2, t1, P, scA, shA, scB, shB, Co);
    }

    // head + top-k
    float* sc = (float*)t1;   // t1 free now; 7 MB of fp32 scores
    score_kernel<<<(unsigned)((size_t)NB * VOX / 256), 256, 0, stream>>>(
        P, (const float*)d_in[28], (const float*)d_in[29], sc);
    topk_init<<<64, 256, 0, stream>>>(prefix, remk, candcnt, hist, cand);
    for (int pass = 0; pass < 4; ++pass) {
        topk_hist<<<dim3(VOX / 4096, NB), 256, 0, stream>>>(sc, prefix, hist, pass);
        topk_select<<<NB, 256, 0, stream>>>(prefix, remk, hist, pass);
    }
    topk_collect<<<dim3(VOX / 4096, NB), 256, 0, stream>>>(sc, prefix, candcnt, cand);
    topk_sort<<<NB, 1024, 0, stream>>>(cand);
    topk_emit<<<dim3(NB, 4), 256, 0, stream>>>(cand, (float*)d_out);
}
